// BiLSTMModel_44470091383025
// MI455X (gfx1250) — matrix operand order repaired and checked by reading, compiled, not run on hardware
//
#include <hip/hip_runtime.h>
#include <hip/hip_bf16.h>
#include <math.h>

// ---------------------------------------------------------------------------
// BiLSTM (B=64,T=512,E=256,H=512,V=512) for MI455X (gfx1250, wave32, WMMA)
//
// Pipeline (all on `stream`):
//   1. fp32->bf16 weight conversion (9 matrices)
//   2. embedding gather -> xs bf16 (T,B,E)
//   3. WMMA GEMM: gates_ih layer0 fwd/bwd  (M=T*B, N=4H, K=E)   [fp32 out]
//      -- W tiles double-buffered in LDS via GLOBAL_LOAD_ASYNC_TO_LDS
//   4. persistent recurrent kernel layer0: fwd+bwd chains concurrently,
//      16 WGs/chain, grid barrier per timestep, whh streamed from L2,
//      h ping-pong through L2, c in registers
//   5. WMMA GEMM: gates_ih layer1 fwd/bwd  (K=H)
//   6. persistent recurrent kernel layer1
//   7. WMMA GEMM: logits = concat(f1, rev(b1)) @ wfc^T + bfc -> d_out fp32
// ---------------------------------------------------------------------------

#if defined(__has_builtin)
#if __has_builtin(__builtin_amdgcn_global_load_async_to_lds_b128) && \
    __has_builtin(__builtin_amdgcn_s_wait_asynccnt)
#define USE_ASYNC_LDS 1
#endif
#endif
#ifndef USE_ASYNC_LDS
#define USE_ASYNC_LDS 0
#endif

namespace {
constexpr int Vv = 512, Ee = 256, Hh = 512, Bb = 64, Tt = 512;
constexpr int G4H = 4 * Hh;            // 2048
constexpr int NWG_CHAIN = 16;          // workgroups per recurrent chain
constexpr int SLICE = Hh / NWG_CHAIN;  // 32 hidden units per WG
static_assert(SLICE == 32, "slice");
}

typedef __attribute__((ext_vector_type(16))) __bf16 bf16x16;
typedef __attribute__((ext_vector_type(8)))  __bf16 bf16x8;
typedef __attribute__((ext_vector_type(8)))  float  f32x8;
typedef __attribute__((ext_vector_type(4)))  int    v4i;

// ---------------------------------------------------------------------------
// device helpers
// ---------------------------------------------------------------------------

__device__ __forceinline__ f32x8 wmma_bf16(bf16x16 a, bf16x16 b, f32x8 c) {
  // D = A(16x32) * B(32x16) + C(16x16 f32)
  return __builtin_amdgcn_wmma_f32_16x16x32_bf16(
      /*neg_a=*/false, a, /*neg_b=*/false, b,
      /*c_mod=*/(short)0, c, /*reuse_a=*/false, /*reuse_b=*/false);
}

// A fragment: lane (half = lane>>4) holds row (lane&15),
//   lanes 0-15: K = 0..15 ; lanes 16-31: K = 16..31 (contiguous per half)
__device__ __forceinline__ bf16x16 load_a16(const __bf16* rowp, int kb, int half) {
  union { bf16x16 v; bf16x8 h[2]; } u;
  const int klo = kb + half * 16;
  u.h[0] = *(const bf16x8*)(rowp + klo);
  u.h[1] = *(const bf16x8*)(rowp + klo + 8);
  return u.v;
}

// B fragment: lane holds weight-row (output column) (lane&15),
//   lanes 0-15: K = 0..15 ; lanes 16-31: K = 16..31 (contiguous per half)
__device__ __forceinline__ bf16x16 load_b16(const __bf16* rowp, int kb, int half) {
  return *(const bf16x16*)(rowp + kb + half * 16);
}

// stage 16 bf16 (32B) from global into LDS; async on CDNA5 when available
__device__ __forceinline__ void stage32B(const __bf16* src, __bf16* lds_dst) {
#if USE_ASYNC_LDS
  typedef __attribute__((address_space(1))) v4i gv4i;  // global int4
  typedef __attribute__((address_space(3))) v4i lv4i;  // LDS int4
  __builtin_amdgcn_global_load_async_to_lds_b128(
      (gv4i*)src, (lv4i*)lds_dst, 0, 0);
  __builtin_amdgcn_global_load_async_to_lds_b128(
      (gv4i*)(src + 8), (lv4i*)(lds_dst + 8), 0, 0);
#else
  *(bf16x8*)lds_dst       = *(const bf16x8*)src;
  *(bf16x8*)(lds_dst + 8) = *(const bf16x8*)(src + 8);
#endif
}

__device__ __forceinline__ void stage_wait() {
#if USE_ASYNC_LDS
  __builtin_amdgcn_s_wait_asynccnt(0);
#endif
}

__device__ __forceinline__ float sigmoidf(float x) {
  return 1.0f / (1.0f + __expf(-x));
}

// sense-reversing grid barrier over gridDim.x workgroups of one chain
__device__ __forceinline__ void chain_barrier(unsigned* bar, unsigned nwg) {
  __syncthreads();
  if (threadIdx.x == 0) {
    __threadfence();
    unsigned gen = __hip_atomic_load(&bar[1], __ATOMIC_RELAXED,
                                     __HIP_MEMORY_SCOPE_AGENT);
    unsigned prev = __hip_atomic_fetch_add(&bar[0], 1u, __ATOMIC_ACQ_REL,
                                           __HIP_MEMORY_SCOPE_AGENT);
    if (prev == nwg - 1u) {
      __hip_atomic_store(&bar[0], 0u, __ATOMIC_RELAXED,
                         __HIP_MEMORY_SCOPE_AGENT);
      __hip_atomic_fetch_add(&bar[1], 1u, __ATOMIC_RELEASE,
                             __HIP_MEMORY_SCOPE_AGENT);
    } else {
      while (__hip_atomic_load(&bar[1], __ATOMIC_ACQUIRE,
                               __HIP_MEMORY_SCOPE_AGENT) == gen) {
        __builtin_amdgcn_s_sleep(2);
      }
    }
  }
  __syncthreads();
  __threadfence();  // acquire: invalidate stale lines before reading peer h
}

// ---------------------------------------------------------------------------
// small kernels
// ---------------------------------------------------------------------------

__global__ void zero_bars(unsigned* p) {
  if (threadIdx.x < 16) p[threadIdx.x] = 0u;
}

__global__ void f32_to_bf16(const float* __restrict__ s,
                            __bf16* __restrict__ d, int n) {
  int i = blockIdx.x * blockDim.x + threadIdx.x;
  if (i < n) d[i] = (__bf16)s[i];
}

// xs[t][b][e] = bf16(emb[x[b][t]][e]);  grid = T*B blocks, E threads
__global__ void embed_bf16(const int* __restrict__ x,
                           const float* __restrict__ emb,
                           __bf16* __restrict__ xs) {
  const int row = blockIdx.x;           // t*B + b
  const int t = row / Bb, b = row - t * Bb;
  const int e = threadIdx.x;            // E == blockDim.x == 256
  const int idx = x[b * Tt + t];
  xs[(size_t)row * Ee + e] = (__bf16)emb[(size_t)idx * Ee + e];
}

// ---------------------------------------------------------------------------
// WMMA GEMM: C[M,N] (fp32) = rowmap(A)[M,K] (bf16) @ W[N,K]^T (bf16) + bias[N]
// Block tile 64(M) x 128(N); 8 waves, wave = 16x64 strip.
// W tiles (128x32 bf16) double-buffered in LDS, streamed with async-to-LDS.
// mode 0: A row m -> m                                 (direct)
// mode 1: m = s*B+b -> (T-1-s)*B+b                     (time reversal)
// mode 2: m = b*T+t -> k<K0: A0[t*B+b], else A1[(T-1-t)*B+b]  (concat fwd/bwd)
// ---------------------------------------------------------------------------

__global__ __launch_bounds__(256) void gemm_bf16_wmma(
    const __bf16* __restrict__ A0, const __bf16* __restrict__ A1,
    const __bf16* __restrict__ W, const float* __restrict__ bias,
    float* __restrict__ C,
    int M, int N, int K, int K0, int Bd, int Td, int mode) {
  const int N0 = blockIdx.x * 128;
  const int M0 = blockIdx.y * 64;
  const int tid = threadIdx.x;
  const int lane = tid & 31, wid = tid >> 5;
  const int half = lane >> 4, li = lane & 15;
  const int mt = wid & 3;   // M sub-tile (16 rows)
  const int ng = wid >> 2;  // owns N sub-tiles ng*4 .. ng*4+3

  __shared__ __bf16 ldsW[2][128 * 48];  // double-buffered 128x32, stride 48

  // per-lane A row pointer(s)
  const int m_lane = M0 + mt * 16 + li;
  const __bf16* rowA0;
  const __bf16* rowA1 = nullptr;
  if (mode == 0) {
    rowA0 = A0 + (size_t)m_lane * K;
  } else if (mode == 1) {
    const int s = m_lane / Bd, b = m_lane - s * Bd;
    rowA0 = A0 + ((size_t)(Td - 1 - s) * Bd + b) * K;
  } else {
    const int b = m_lane / Td, t = m_lane - b * Td;
    rowA0 = A0 + ((size_t)t * Bd + b) * K0;
    rowA1 = A1 + ((size_t)(Td - 1 - t) * Bd + b) * (K - K0);
  }

  // W staging: 256 threads cover 128 rows x 2 half-chunks of 16 bf16
  const int srow = tid >> 1, sh = tid & 1;
  const __bf16* wsrc = W + (size_t)(N0 + srow) * K + sh * 16;
  const int ldst = srow * 48 + sh * 16;

  // prologue: stage first tile
  stage32B(wsrc, &ldsW[0][ldst]);
  stage_wait();
  __syncthreads();

  f32x8 acc[4] = {};
  for (int kb = 0; kb < K; kb += 32) {
    const int cur = (kb >> 5) & 1;
    const int nxt = cur ^ 1;
    const bool more = (kb + 32) < K;
    if (more) stage32B(wsrc + kb + 32, &ldsW[nxt][ldst]);  // overlap with math

    const __bf16* ra = rowA0;
    int kk = kb;
    if (mode == 2 && kb >= K0) { ra = rowA1; kk = kb - K0; }
    const bf16x16 a = load_a16(ra, kk, half);

#pragma unroll
    for (int j = 0; j < 4; ++j) {
      const int nrow = (ng * 4 + j) * 16 + li;
      const bf16x16 w = *(const bf16x16*)(&ldsW[cur][nrow * 48 + half * 16]);
      acc[j] = wmma_bf16(a, w, acc[j]);
    }

    if (more) stage_wait();  // next tile resident before buffer swap
    __syncthreads();
  }

#pragma unroll
  for (int j = 0; j < 4; ++j) {
    const int n = N0 + (ng * 4 + j) * 16 + li;
    const float bn = bias ? bias[n] : 0.0f;
#pragma unroll
    for (int r = 0; r < 8; ++r) {
      const int m = M0 + mt * 16 + r + 8 * half;
      C[(size_t)m * N + n] = acc[j][r] + bn;
    }
  }
}

// ---------------------------------------------------------------------------
// persistent LSTM recurrence: grid (NWG_CHAIN, 2 chains), 256 threads.
// Each WG owns SLICE=32 hidden units -> 128 whh rows (i,f,g,o blocks).
// Per step: gates_hh = h @ whh_slice^T via WMMA (M=64,N=128,K=512),
// exchange through padded LDS, elementwise gate math, c in registers,
// h ping-pong in global, grid barrier per step.
// ---------------------------------------------------------------------------

__global__ __launch_bounds__(256) void lstm_recurrent(
    const float* __restrict__ gihF, const float* __restrict__ gihB,
    const __bf16* __restrict__ whhF, const __bf16* __restrict__ whhB,
    __bf16* __restrict__ hseqF, __bf16* __restrict__ hseqB,
    __bf16* __restrict__ hwF, __bf16* __restrict__ hwB,
    unsigned* __restrict__ bars) {
  const int chain = blockIdx.y;
  const float*  gih  = chain ? gihB : gihF;
  const __bf16* whh  = chain ? whhB : whhF;
  __bf16*       hseq = chain ? hseqB : hseqF;
  __bf16*       hw   = chain ? hwB : hwF;
  unsigned*     bar  = bars + chain * 8;

  const int wg = blockIdx.x;
  const int js = wg * SLICE;  // hidden slice base
  const int tid = threadIdx.x;
  const int lane = tid & 31, wid = tid >> 5;
  const int half = lane >> 4, li = lane & 15;
  const int mt = wid & 3;   // batch tile
  const int ng = wid >> 2;  // N-tile group

  __shared__ float gbuf[128 * 65];  // gates_hh[n][m], stride 65 (pad)

  // per-lane whh row pointers for this wave's 4 N-tiles
  const __bf16* wrow[4];
#pragma unroll
  for (int j = 0; j < 4; ++j) {
    const int nl = (ng * 4 + j) * 16 + li;  // 0..127
    const int gate = nl >> 5;
    const int jj = js + (nl & 31);
    wrow[j] = whh + ((size_t)(gate * Hh + jj)) * Hh;
  }
  const int arow = mt * 16 + li;  // batch row

  __bf16* hbuf0 = hw;
  __bf16* hbuf1 = hw + Bb * Hh;

  // zero this WG's slice of the t=0 read buffer; c starts at 0
  float c[8];
#pragma unroll
  for (int q = 0; q < 8; ++q) {
    const int p = q * 256 + tid;  // 0..2047 = (b, jl)
    hbuf0[(p >> 5) * Hh + js + (p & 31)] = (__bf16)0.0f;
    c[q] = 0.0f;
  }
  __threadfence();
  chain_barrier(bar, gridDim.x);

  for (int t = 0; t < Tt; ++t) {
    const __bf16* hin = (t & 1) ? hbuf1 : hbuf0;
    __bf16*      hout = (t & 1) ? hbuf0 : hbuf1;
    const __bf16* ap = hin + (size_t)arow * Hh;

    // pull this step's fp32 ih-gate lines toward the WGP while WMMAs run
#pragma unroll
    for (int q = 0; q < 8; ++q) {
      const int p = q * 256 + tid;
      const int b = p >> 5, jl = p & 31;
      __builtin_prefetch(gih + ((size_t)t * Bb + b) * G4H + js + jl, 0, 1);
    }

    f32x8 acc[4] = {};
#pragma unroll 4
    for (int ks = 0; ks < Hh / 32; ++ks) {
      const int kb = ks * 32;
      const bf16x16 a = load_a16(ap, kb, half);
#pragma unroll
      for (int j = 0; j < 4; ++j) {
        const bf16x16 w = load_b16(wrow[j], kb, half);
        acc[j] = wmma_bf16(a, w, acc[j]);
      }
    }

    // dump hh sums to LDS (padded to avoid bank conflicts)
#pragma unroll
    for (int j = 0; j < 4; ++j) {
      const int nl = (ng * 4 + j) * 16 + li;
#pragma unroll
      for (int r = 0; r < 8; ++r) {
        const int m = mt * 16 + r + 8 * half;
        gbuf[nl * 65 + m] = acc[j][r];
      }
    }
    __syncthreads();

    // gate math + state update; thread owns 8 (b, jl) pairs
#pragma unroll
    for (int q = 0; q < 8; ++q) {
      const int p = q * 256 + tid;
      const int b = p >> 5, jl = p & 31;
      const float* grow = gih + ((size_t)t * Bb + b) * G4H + js + jl;
      const float xi = gbuf[(0 * 32 + jl) * 65 + b] + grow[0 * Hh];
      const float xf = gbuf[(1 * 32 + jl) * 65 + b] + grow[1 * Hh];
      const float xg = gbuf[(2 * 32 + jl) * 65 + b] + grow[2 * Hh];
      const float xo = gbuf[(3 * 32 + jl) * 65 + b] + grow[3 * Hh];
      const float ig = sigmoidf(xi), fg = sigmoidf(xf);
      const float gg = tanhf(xg),    og = sigmoidf(xo);
      c[q] = fg * c[q] + ig * gg;
      const float hn = og * tanhf(c[q]);
      const __bf16 hb = (__bf16)hn;
      hout[b * Hh + js + jl] = hb;
      hseq[((size_t)t * Bb + b) * Hh + js + jl] = hb;
    }
    __threadfence();               // publish hout
    chain_barrier(bar, gridDim.x); // also guards gbuf reuse
  }
}

// ---------------------------------------------------------------------------
// host: launch pipeline
// ---------------------------------------------------------------------------

extern "C" void kernel_launch(void* const* d_in, const int* in_sizes, int n_in,
                              void* d_out, int out_size, void* d_ws,
                              size_t ws_size, hipStream_t stream) {
  (void)in_sizes; (void)n_in; (void)out_size; (void)ws_size;

  const int*   x      = (const int*)  d_in[0];
  const float* emb    = (const float*)d_in[1];
  const float* wih_f0 = (const float*)d_in[2];
  const float* whh_f0 = (const float*)d_in[3];
  const float* b_f0   = (const float*)d_in[4];
  const float* wih_f1 = (const float*)d_in[5];
  const float* whh_f1 = (const float*)d_in[6];
  const float* b_f1   = (const float*)d_in[7];
  const float* wih_b0 = (const float*)d_in[8];
  const float* whh_b0 = (const float*)d_in[9];
  const float* b_b0   = (const float*)d_in[10];
  const float* wih_b1 = (const float*)d_in[11];
  const float* whh_b1 = (const float*)d_in[12];
  const float* b_b1   = (const float*)d_in[13];
  const float* wfc    = (const float*)d_in[14];
  const float* bfc    = (const float*)d_in[15];
  float* out = (float*)d_out;

  char* ws = (char*)d_ws;
  size_t off = 0;
  auto take = [&](size_t bytes) -> char* {
    char* p = ws + off;
    off = (off + bytes + 255) & ~(size_t)255;
    return p;
  };

  __bf16* xs      = (__bf16*)take((size_t)Tt * Bb * Ee * 2);
  __bf16* wihf0b  = (__bf16*)take((size_t)G4H * Ee * 2);
  __bf16* whhf0b  = (__bf16*)take((size_t)G4H * Hh * 2);
  __bf16* wihf1b  = (__bf16*)take((size_t)G4H * Hh * 2);
  __bf16* whhf1b  = (__bf16*)take((size_t)G4H * Hh * 2);
  __bf16* wihb0b  = (__bf16*)take((size_t)G4H * Ee * 2);
  __bf16* whhb0b  = (__bf16*)take((size_t)G4H * Hh * 2);
  __bf16* wihb1b  = (__bf16*)take((size_t)G4H * Hh * 2);
  __bf16* whhb1b  = (__bf16*)take((size_t)G4H * Hh * 2);
  __bf16* wfcb    = (__bf16*)take((size_t)Vv * 2 * Hh * 2);
  float*  gihF    = (float*) take((size_t)Tt * Bb * G4H * 4);
  float*  gihB    = (float*) take((size_t)Tt * Bb * G4H * 4);
  __bf16* hseqf0  = (__bf16*)take((size_t)Tt * Bb * Hh * 2);
  __bf16* hseqb0  = (__bf16*)take((size_t)Tt * Bb * Hh * 2);
  __bf16* hseqf1  = (__bf16*)take((size_t)Tt * Bb * Hh * 2);
  __bf16* hseqb1  = (__bf16*)take((size_t)Tt * Bb * Hh * 2);
  __bf16* hwF     = (__bf16*)take((size_t)2 * Bb * Hh * 2);
  __bf16* hwB     = (__bf16*)take((size_t)2 * Bb * Hh * 2);
  unsigned* bars  = (unsigned*)take(64);

  const dim3 blk(256);
  auto cvt = [&](const float* s, __bf16* d, int n) {
    f32_to_bf16<<<dim3((n + 255) / 256), blk, 0, stream>>>(s, d, n);
  };
  cvt(wih_f0, wihf0b, G4H * Ee);
  cvt(whh_f0, whhf0b, G4H * Hh);
  cvt(wih_f1, wihf1b, G4H * Hh);
  cvt(whh_f1, whhf1b, G4H * Hh);
  cvt(wih_b0, wihb0b, G4H * Ee);
  cvt(whh_b0, whhb0b, G4H * Hh);
  cvt(wih_b1, wihb1b, G4H * Hh);
  cvt(whh_b1, whhb1b, G4H * Hh);
  cvt(wfc,    wfcb,   Vv * 2 * Hh);

  embed_bf16<<<dim3(Tt * Bb), dim3(Ee), 0, stream>>>(x, emb, xs);

  const dim3 gih_grid(G4H / 128, (Tt * Bb) / 64);
  // layer-0 input projections (forward: direct, backward: time-reversed)
  gemm_bf16_wmma<<<gih_grid, blk, 0, stream>>>(
      xs, nullptr, wihf0b, b_f0, gihF, Tt * Bb, G4H, Ee, Ee, Bb, Tt, 0);
  gemm_bf16_wmma<<<gih_grid, blk, 0, stream>>>(
      xs, nullptr, wihb0b, b_b0, gihB, Tt * Bb, G4H, Ee, Ee, Bb, Tt, 1);

  zero_bars<<<1, 32, 0, stream>>>(bars);
  lstm_recurrent<<<dim3(NWG_CHAIN, 2), blk, 0, stream>>>(
      gihF, gihB, whhf0b, whhb0b, hseqf0, hseqb0, hwF, hwB, bars);

  // layer-1 input projections (both direct: sequences already chain-ordered)
  gemm_bf16_wmma<<<gih_grid, blk, 0, stream>>>(
      hseqf0, nullptr, wihf1b, b_f1, gihF, Tt * Bb, G4H, Hh, Hh, Bb, Tt, 0);
  gemm_bf16_wmma<<<gih_grid, blk, 0, stream>>>(
      hseqb0, nullptr, wihb1b, b_b1, gihB, Tt * Bb, G4H, Hh, Hh, Bb, Tt, 0);

  zero_bars<<<1, 32, 0, stream>>>(bars);
  lstm_recurrent<<<dim3(NWG_CHAIN, 2), blk, 0, stream>>>(
      gihF, gihB, whhf1b, whhb1b, hseqf1, hseqb1, hwF, hwB, bars);

  // logits[b,t,:] = [f1[t,b,:], b1[T-1-t,b,:]] @ wfc^T + bfc
  const dim3 fc_grid(Vv / 128, (Bb * Tt) / 64);
  gemm_bf16_wmma<<<fc_grid, blk, 0, stream>>>(
      hseqf1, hseqb1, wfcb, bfc, out, Bb * Tt, Vv, 2 * Hh, Hh, Bb, Tt, 2);
}
